// GATv2ConvLayer_35107062678099
// MI455X (gfx1250) — matrix operand order repaired and checked by reading, not compiled, not run on hardware
//
#include <hip/hip_runtime.h>
#include <hip/hip_bf16.h>
#include <math.h>

// ---------------------------------------------------------------------------
// GATv2 layer for MI455X (gfx1250).  Dense contractions use
// v_wmma_f32_16x16x32_bf16; the edge-feature intermediate is stored bf16 and
// staged into LDS with global_load_async_to_lds_b128 (ASYNCcnt path);
// irregular gather/softmax/scatter paths use VALU + L2 f32 atomics.
// ---------------------------------------------------------------------------

#define IN_CH     256
#define HC        256          // H*C
#define ED        64
#define NHEAD     8
#define CPH       32
#define EA_STRIDE 65           // edge_attr row = [type, 64 feats]
#define NEG_SLOPE 0.2f

typedef __attribute__((ext_vector_type(16))) __bf16 v16bf;
typedef __attribute__((ext_vector_type(8)))  float  v8f;

// ---------------- WMMA helpers (CDNA5 16x16x32 bf16) -----------------------

__device__ __forceinline__ v8f wmma_bf16(v16bf a, v16bf b, v8f c) {
  // 8 args: (neg_a, A, neg_b, B, c_mod, C, reuse_a, reuse_b)
  return __builtin_amdgcn_wmma_f32_16x16x32_bf16(false, a, false, b,
                                                 (short)0, c, false, false);
}

// A fragment: 16(M) x 32(K) bf16.  Lane L<16 holds row M=L, elements
// 0..7 -> K 0..7, 8..15 -> K 16..23; lanes 16..31 hold K+8 offsets.
__device__ __forceinline__ v16bf load_a_frag(const __bf16* tile, int ld,
                                             int kbase, int lane) {
  const int m  = lane & 15;
  const int hi = lane >> 4;
  const __bf16* row = tile + (size_t)m * ld + kbase + hi * 8;
  v16bf a;
#pragma unroll
  for (int e = 0; e < 16; ++e) {
    const int k = (e < 8) ? e : (e + 8);
    a[e] = row[k];
  }
  return a;
}

// B fragment: 32(K) x 16(N) bf16.  Lane L holds column N=L&15,
// elements 0..7 -> K 0..7, 8..15 -> K 16..23; lanes 16..31 hold K+8 offsets.
__device__ __forceinline__ v16bf load_b_frag(const __bf16* tile, int ld,
                                             int kbase, int nbase, int lane) {
  const __bf16* col = tile + (size_t)(kbase + (lane >> 4) * 8) * ld + nbase + (lane & 15);
  v16bf b;
#pragma unroll
  for (int e = 0; e < 16; ++e) b[e] = col[(size_t)((e < 8) ? e : (e + 8)) * ld];
  return b;
}

// C/D: lane L, register r -> (M = r + 8*(L>>4), N = L&15)

// ---------------- float atomic max via ordered-uint encoding ---------------

__device__ __forceinline__ unsigned fenc(float x) {
  unsigned u = __float_as_uint(x);
  return (u & 0x80000000u) ? ~u : (u | 0x80000000u);
}
__device__ __forceinline__ float fdec(unsigned u) {
  return (u & 0x80000000u) ? __uint_as_float(u ^ 0x80000000u)
                           : __uint_as_float(~u);
}

// ---------------- init kernels --------------------------------------------

__global__ void k_init_amax(unsigned* amax, int n) {
  int i = blockIdx.x * blockDim.x + threadIdx.x;
  if (i < n) amax[i] = 0x007FFFFFu;  // fenc(-inf)
}

__global__ void k_init_out(float* out, const float* __restrict__ bias, size_t n) {
  size_t i = (size_t)blockIdx.x * blockDim.x + threadIdx.x;
  if (i < n) out[i] = bias[i & (HC - 1)];
}

// ---------------- K1: edge encoder MLP (two 64x64 WMMA layers) -------------
// eab[e] = relu(ef @ We1^T + be1) @ We2^T + be2 (stored bf16),
// ef = efeat + type_emb[et].  Also accumulates f32 per-dst sums (loopsum)
// for the self-loop 'mean' fill attribute.

__global__ void __launch_bounds__(256)
k_edge_mlp(const float* __restrict__ eattr, const float* __restrict__ type_emb,
           const float* __restrict__ We1, const float* __restrict__ be1,
           const float* __restrict__ We2, const float* __restrict__ be2,
           const int* __restrict__ dst0, __bf16* __restrict__ eab,
           float* __restrict__ loopsum, float* __restrict__ cnt, int E) {
  __shared__ __bf16 WB1[ED][ED + 8];      // B layout: [k][n] = We1[n][k]
  __shared__ __bf16 WB2[ED][ED + 8];
  __shared__ __bf16 ef[8][16][ED + 8];    // per-wave A tiles
  __shared__ __bf16 hh[8][16][ED + 8];

  const int t    = threadIdx.x;
  const int lane = t & 31;
  const int w    = t >> 5;

  // cooperative load + transpose of the two 64x64 weight matrices
#pragma unroll
  for (int j = 0; j < 16; ++j) {
    int idx = t + 256 * j;              // idx < 4096
    int o = idx >> 6, d = idx & 63;     // global read coalesced along d
    WB1[d][o] = (__bf16)We1[idx];
    WB2[d][o] = (__bf16)We2[idx];
  }
  __syncthreads();

  const int ebase = (blockIdx.x * 8 + w) * 16;
  if (ebase >= E) return;               // E is a multiple of 16

  // stage ef = efeat + type_emb[et] as bf16
  for (int idx = lane; idx < 16 * ED; idx += 32) {
    int m = idx >> 6, d = idx & 63;
    const float* row = eattr + (size_t)(ebase + m) * EA_STRIDE;
    int et = (int)row[0];
    ef[w][m][d] = (__bf16)(row[1 + d] + type_emb[et * ED + d]);
  }

  const int hi = lane >> 4;
  const int nl = lane & 15;

  // layer 1: h = relu(ef @ We1^T + be1)
#pragma unroll
  for (int nt = 0; nt < 4; ++nt) {
    v8f acc = {};
#pragma unroll
    for (int ks = 0; ks < 2; ++ks) {
      v16bf a = load_a_frag(&ef[w][0][0], ED + 8, ks * 32, lane);
      v16bf b = load_b_frag(&WB1[0][0], ED + 8, ks * 32, nt * 16, lane);
      acc = wmma_bf16(a, b, acc);
    }
    const int n = nt * 16 + nl;
    const float bn = be1[n];
#pragma unroll
    for (int r = 0; r < 8; ++r) {
      float v = acc[r] + bn;
      hh[w][r + 8 * hi][n] = (__bf16)(v > 0.f ? v : 0.f);
    }
  }

  // layer 2: ea = h @ We2^T + be2 ; bf16 write + f32 atomic per-dst sums
#pragma unroll
  for (int nt = 0; nt < 4; ++nt) {
    v8f acc = {};
#pragma unroll
    for (int ks = 0; ks < 2; ++ks) {
      v16bf a = load_a_frag(&hh[w][0][0], ED + 8, ks * 32, lane);
      v16bf b = load_b_frag(&WB2[0][0], ED + 8, ks * 32, nt * 16, lane);
      acc = wmma_bf16(a, b, acc);
    }
    const int n = nt * 16 + nl;
    const float bn = be2[n];
#pragma unroll
    for (int r = 0; r < 8; ++r) {
      int e = ebase + r + 8 * hi;
      float v = acc[r] + bn;
      eab[(size_t)e * ED + n] = (__bf16)v;
      atomicAdd(&loopsum[(size_t)dst0[e] * ED + n], v);
    }
  }
  if (lane < 16) atomicAdd(&cnt[dst0[ebase + lane]], 1.0f);
}

// ---------------- K2: loop_attr rows = sums / max(cnt,1), stored bf16 ------

__global__ void k_loop_div(__bf16* __restrict__ eab,
                           const float* __restrict__ loopsum,
                           const float* __restrict__ cnt, int E, int N) {
  int i = blockIdx.x * blockDim.x + threadIdx.x;
  if (i >= N * ED) return;
  int n = i >> 6;
  eab[(size_t)(E + n) * ED + (i & 63)] =
      (__bf16)(loopsum[i] * (1.0f / fmaxf(cnt[n], 1.0f)));
}

// ---------------- K3: node transform GEMM  OUT = X @ W^T + b ---------------
// M x 256 @ (256 x 256)^T, 128x64 block tile, 8 waves x (16x64).

__global__ void __launch_bounds__(256)
k_gemm256(const float* __restrict__ X, const float* __restrict__ W,
          const float* __restrict__ bias, float* __restrict__ OUT, int M) {
  __shared__ __bf16 As[128][32 + 8];
  __shared__ __bf16 Bs[32][64 + 8];

  const int t = threadIdx.x, lane = t & 31, w = t >> 5;
  const int rowbase = blockIdx.x * 128;
  const int nbase   = blockIdx.y * 64;
  const int hi = lane >> 4, nl = lane & 15;

  v8f acc[4] = {{}, {}, {}, {}};

  for (int kb = 0; kb < IN_CH; kb += 32) {
#pragma unroll
    for (int j = 0; j < 16; ++j) {                  // A: 128x32
      int idx = t + 256 * j;
      int m = idx >> 5, k = idx & 31, r = rowbase + m;
      As[m][k] = (r < M) ? (__bf16)X[(size_t)r * IN_CH + kb + k] : (__bf16)0.f;
    }
#pragma unroll
    for (int j = 0; j < 8; ++j) {                   // B: [k][n] = W[n][k]
      int idx = t + 256 * j;
      int k = idx & 31, n = idx >> 5;
      Bs[k][n] = (__bf16)W[(size_t)(nbase + n) * IN_CH + kb + k];
    }
    __syncthreads();
    v16bf a = load_a_frag(&As[16 * w][0], 32 + 8, 0, lane);
#pragma unroll
    for (int nt = 0; nt < 4; ++nt) {
      v16bf b = load_b_frag(&Bs[0][0], 64 + 8, 0, nt * 16, lane);
      acc[nt] = wmma_bf16(a, b, acc[nt]);
    }
    __syncthreads();
  }
#pragma unroll
  for (int nt = 0; nt < 4; ++nt) {
    int n = nbase + nt * 16 + nl;
    float bn = bias[n];
#pragma unroll
    for (int r = 0; r < 8; ++r) {
      int row = rowbase + 16 * w + r + 8 * hi;
      if (row < M) OUT[(size_t)row * HC + n] = acc[nt][r] + bn;
    }
  }
}

// ---------------- K4: attention scores -------------------------------------
// e_feat = ea @ Wedge^T via WMMA (16 edges x 256 per wave-tile), then
// z = leaky_relu(xl[src] + xr[dst] + e_feat), logits[e,h] = z . att[h].
// The bf16 ea tile is staged global->LDS with the CDNA5 async copy path.

__global__ void __launch_bounds__(256)
k_score(const __bf16* __restrict__ eab, const float* __restrict__ Wedge,
        const float* __restrict__ att, const float* __restrict__ xl,
        const float* __restrict__ xr, const int* __restrict__ eidx,
        float* __restrict__ logits, unsigned* __restrict__ amax,
        int E, int EN) {
  __shared__ __bf16 WB[ED][HC + 8];          // [k][n] = Wedge[n][k]
  __shared__ __bf16 eaT[8][16][ED + 8];      // row stride 144B (16B aligned)
  __shared__ float  attS[HC];

  const int t = threadIdx.x, lane = t & 31, w = t >> 5;
  const int* src0 = eidx;
  const int* dst0 = eidx + E;

  const int ebase = (blockIdx.x * 8 + w) * 16;

  // --- async-copy the 16x64 bf16 ea tile (2KB) into LDS, honoring row pad.
  // Each GLOBAL_LOAD_ASYNC_TO_LDS_B128 moves 16B/lane = 512B/wave = 4 rows.
  if (ebase < EN) {
    unsigned ldsbase = (unsigned)(uintptr_t)(void*)&eaT[w][0][0];
    unsigned long long gbase =
        (unsigned long long)(uintptr_t)(const void*)(eab + (size_t)ebase * ED)
        + (unsigned long long)lane * 16ull;
    unsigned loff0 = ldsbase + (unsigned)(lane >> 3) * ((ED + 8) * 2)
                             + (unsigned)(lane & 7) * 16u;
#pragma unroll
    for (int j = 0; j < 4; ++j) {
      unsigned loff = loff0 + (unsigned)j * 4u * ((ED + 8) * 2);
      unsigned long long ga = gbase + (unsigned long long)j * 512ull;
      asm volatile("global_load_async_to_lds_b128 %0, %1, off"
                   :: "v"(loff), "v"(ga) : "memory");
    }
  }

  // --- meanwhile fill Wedge^T and att into LDS (block cooperative)
#pragma unroll
  for (int j = 0; j < 64; ++j) {             // Wedge: 256x64
    int idx = t + 256 * j;
    int n = idx >> 6, k = idx & 63;
    WB[k][n] = (__bf16)Wedge[idx];
  }
  attS[t] = att[t];
  __syncthreads();

  if (ebase >= EN) return;                   // EN is a multiple of 16

  const int hi = lane >> 4, nl = lane & 15;
  int sArr[8], dArr[8];
#pragma unroll
  for (int r = 0; r < 8; ++r) {
    int e = ebase + r + 8 * hi;
    if (e < E) { sArr[r] = src0[e]; dArr[r] = dst0[e]; }
    else       { sArr[r] = dArr[r] = e - E; }       // self loop
  }

  // wait for this wave's async tile before building fragments
  asm volatile("s_wait_asynccnt 0x0" ::: "memory");

  v16bf a0 = load_a_frag(&eaT[w][0][0], ED + 8, 0, lane);
  v16bf a1 = load_a_frag(&eaT[w][0][0], ED + 8, 32, lane);

  for (int h = 0; h < NHEAD; ++h) {
    float sacc[8] = {0, 0, 0, 0, 0, 0, 0, 0};
#pragma unroll
    for (int sub = 0; sub < 2; ++sub) {
      const int nb = (2 * h + sub) * 16;
      v8f acc = {};
      acc = wmma_bf16(a0, load_b_frag(&WB[0][0], HC + 8, 0, nb, lane), acc);
      acc = wmma_bf16(a1, load_b_frag(&WB[0][0], HC + 8, 32, nb, lane), acc);
      const int n = nb + nl;
      const float an = attS[n];
#pragma unroll
      for (int r = 0; r < 8; ++r) {
        float v = acc[r] + xl[(size_t)sArr[r] * HC + n]
                         + xr[(size_t)dArr[r] * HC + n];
        v = (v > 0.f) ? v : NEG_SLOPE * v;
        sacc[r] += v * an;
      }
    }
    // reduce across the 16 lanes of each half (N dimension)
#pragma unroll
    for (int r = 0; r < 8; ++r) {
      float s = sacc[r];
      s += __shfl_xor(s, 1, 32);
      s += __shfl_xor(s, 2, 32);
      s += __shfl_xor(s, 4, 32);
      s += __shfl_xor(s, 8, 32);
      sacc[r] = s;
    }
    if (nl == 0) {
#pragma unroll
      for (int r = 0; r < 8; ++r) {
        int e = ebase + r + 8 * hi;
        logits[(size_t)e * NHEAD + h] = sacc[r];
        atomicMax(&amax[(size_t)dArr[r] * NHEAD + h], fenc(sacc[r]));
      }
    }
  }
}

// ---------------- K5: softmax denominators ---------------------------------

__global__ void k_denom(const float* __restrict__ logits,
                        const unsigned* __restrict__ amax,
                        const int* __restrict__ eidx,
                        float* __restrict__ denom, int E, int EN) {
  int i = blockIdx.x * blockDim.x + threadIdx.x;
  if (i >= EN * NHEAD) return;
  int e = i >> 3, h = i & 7;
  int d = (e < E) ? eidx[E + e] : (e - E);
  float ex = __expf(logits[i] - fdec(amax[(size_t)d * NHEAD + h]));
  atomicAdd(&denom[(size_t)d * NHEAD + h], ex);
}

// ---------------- K6: weighted scatter-aggregate ---------------------------

__global__ void __launch_bounds__(256)
k_aggregate(const float* __restrict__ xl, const float* __restrict__ logits,
            const unsigned* __restrict__ amax, const float* __restrict__ denom,
            const int* __restrict__ eidx, float* __restrict__ out,
            int E, int EN) {
  const int e = blockIdx.x;
  const int c = threadIdx.x;
  int s, d;
  if (e < E) { s = eidx[e]; d = eidx[E + e]; }
  else       { s = d = e - E; }
  const int h = c >> 5;
  float alpha = __expf(logits[(size_t)e * NHEAD + h]
                       - fdec(amax[(size_t)d * NHEAD + h]))
                / (denom[(size_t)d * NHEAD + h] + 1e-16f);
  atomicAdd(&out[(size_t)d * HC + c], xl[(size_t)s * HC + c] * alpha);
}

// ---------------------------------------------------------------------------

extern "C" void kernel_launch(void* const* d_in, const int* in_sizes, int n_in,
                              void* d_out, int out_size, void* d_ws, size_t ws_size,
                              hipStream_t stream) {
  const float* x        = (const float*)d_in[0];
  const int*   eidx     = (const int*)d_in[1];      // [2,E] src row then dst row
  const float* eattr    = (const float*)d_in[2];    // [E, 65]
  const float* type_emb = (const float*)d_in[3];
  const float* We1      = (const float*)d_in[4];
  const float* be1      = (const float*)d_in[5];
  const float* We2      = (const float*)d_in[6];
  const float* be2      = (const float*)d_in[7];
  const float* Wl       = (const float*)d_in[8];
  const float* bl       = (const float*)d_in[9];
  const float* Wr       = (const float*)d_in[10];
  const float* br       = (const float*)d_in[11];
  const float* Wedge    = (const float*)d_in[12];
  const float* att      = (const float*)d_in[13];
  const float* bias     = (const float*)d_in[14];

  const int N  = in_sizes[0] / IN_CH;
  const int E  = in_sizes[1] / 2;
  const int EN = E + N;

  // workspace carve-out (4-byte units)
  float* ws = (float*)d_ws;
  size_t o = 0;
  __bf16*   eab     = (__bf16*)(ws + o); o += (size_t)EN * ED / 2;  // bf16
  float*    loopsum = ws + o; o += (size_t)N * ED;
  float*    xl      = ws + o; o += (size_t)N * HC;
  float*    xr      = ws + o; o += (size_t)N * HC;
  float*    logits  = ws + o; o += (size_t)EN * NHEAD;
  unsigned* amax    = (unsigned*)(ws + o); o += (size_t)N * NHEAD;
  float*    denom   = ws + o; o += (size_t)N * NHEAD;
  float*    cnt     = ws + o; o += (size_t)N;

  // per-call re-initialization (graph-capture safe)
  hipMemsetAsync(loopsum, 0, (size_t)N * ED * sizeof(float), stream);
  hipMemsetAsync(cnt,     0, (size_t)N * sizeof(float), stream);
  hipMemsetAsync(denom,   0, (size_t)N * NHEAD * sizeof(float), stream);
  k_init_amax<<<(N * NHEAD + 255) / 256, 256, 0, stream>>>(amax, N * NHEAD);
  k_init_out<<<(int)(((size_t)N * HC + 255) / 256), 256, 0, stream>>>(
      (float*)d_out, bias, (size_t)N * HC);

  // 1) edge encoder MLP (WMMA) + self-loop mean accumulation
  k_edge_mlp<<<(E + 127) / 128, 256, 0, stream>>>(
      eattr, type_emb, We1, be1, We2, be2, eidx + E, eab, loopsum, cnt, E);
  k_loop_div<<<(N * ED + 255) / 256, 256, 0, stream>>>(eab, loopsum, cnt, E, N);

  // 2) node transforms (WMMA GEMMs)
  dim3 ggrid((N + 127) / 128, HC / 64);
  k_gemm256<<<ggrid, 256, 0, stream>>>(x, Wl, bl, xl, N);
  k_gemm256<<<ggrid, 256, 0, stream>>>(x, Wr, br, xr, N);

  // 3) attention logits (WMMA e_feat + gathered node terms) + segment max
  k_score<<<(EN + 127) / 128, 256, 0, stream>>>(
      eab, Wedge, att, xl, xr, eidx, logits, amax, E, EN);

  // 4) softmax denominators
  k_denom<<<(EN * NHEAD + 255) / 256, 256, 0, stream>>>(
      logits, amax, eidx, denom, E, EN);

  // 5) scatter aggregation
  k_aggregate<<<EN, 256, 0, stream>>>(
      xl, logits, amax, denom, eidx, (float*)d_out, E, EN);
}
